// EdgeMP_84894323573138
// MI455X (gfx1250) — matrix operand order repaired and checked by reading, not compiled, not run on hardware
//
#include <hip/hip_runtime.h>
#include <hip/hip_bf16.h>
#include <math.h>

#define NN 50000
#define NE 800000
#define DM 64
#define DE 32
#define DH 256
#define NG 64
#define EPSV 1e-5f

typedef __bf16 bf16_t;
typedef bf16_t v16bf __attribute__((ext_vector_type(16)));
typedef float v8f __attribute__((ext_vector_type(8)));
typedef unsigned short v8us __attribute__((ext_vector_type(8)));
typedef unsigned short us4 __attribute__((ext_vector_type(4)));

union Frag {
    v16bf bf;
    v8us  us[2];
};

__device__ __forceinline__ unsigned short f2bf(float f) {
    unsigned int u = __float_as_uint(f);
    u += 0x7FFFu + ((u >> 16) & 1u);   // round-to-nearest-even
    return (unsigned short)(u >> 16);
}

// fast SiLU: x * rcp(1 + exp(-x))  -> v_exp_f32 + v_rcp_f32 + v_mul
__device__ __forceinline__ float silu(float x) {
    return x * __builtin_amdgcn_rcpf(1.0f + __expf(-x));
}

__device__ __forceinline__ v8f wmma_bf16(v16bf a, v16bf b, v8f c) {
    return __builtin_amdgcn_wmma_f32_16x16x32_bf16(false, a, false, b, (short)0, c, false, false);
}

// A fragment (16x32 bf16 M-tile) from a row-major bf16 tile in LDS.
// lanes 0-15 -> M=lane,   K = k0+0..7  and k0+16..23
// lanes 16-31 -> M=lane-16, K = k0+8..15 and k0+24..31
__device__ __forceinline__ v16bf load_a_frag(const unsigned short* rowbase, int stride, int lane) {
    int row = lane & 15;
    const unsigned short* p = rowbase + row * stride + ((lane < 16) ? 0 : 8);
    Frag f;
    f.us[0] = *(const v8us*)(p);
    f.us[1] = *(const v8us*)(p + 16);
    return f.bf;
}

// B fragment from pre-packed weights: per (kt,nt) tile, 32 lanes x 16 bf16 contiguous.
__device__ __forceinline__ v16bf load_b_frag(const unsigned short* wp, int kt, int nt, int NT, int lane) {
    const unsigned short* p = wp + ((size_t)((kt * NT + nt) * 32 + lane)) * 16;
    Frag f;
    f.us[0] = *(const v8us*)(p);
    f.us[1] = *(const v8us*)(p + 8);
    return f.bf;
}

// ---------------- weight packing: W[K,N] fp32 row-major -> WMMA B-fragment bf16 layout
__global__ void pack_weights(const float* __restrict__ W, int K, int N,
                             unsigned short* __restrict__ out) {
    int t = blockIdx.x * blockDim.x + threadIdx.x;
    int NT = N >> 4;
    int KT = K >> 5;
    if (t >= KT * NT * 32) return;
    int lane = t & 31;
    int nt = (t >> 5) % NT;
    int kt = t / (32 * NT);
    int n  = nt * 16 + (lane & 15);
    int kb = kt * 32 + ((lane < 16) ? 0 : 8);
    unsigned short* o = out + (size_t)t * 16;
#pragma unroll
    for (int j = 0; j < 16; ++j)
        o[j] = f2bf(W[(size_t)(kb + (j < 8 ? j : j + 8)) * N + n]);
}

// ---------------- zero scratch accumulators
__global__ void zero_ws(float* __restrict__ agg, float* __restrict__ sums,
                        float* __restrict__ vars, float* __restrict__ cnts) {
    int stride = gridDim.x * blockDim.x;
    int t = blockIdx.x * blockDim.x + threadIdx.x;
    for (int i = t; i < NN * DM; i += stride) agg[i] = 0.0f;
    if (t < NG * DM) { sums[t] = 0.0f; vars[t] = 0.0f; }
    if (t < NG) cnts[t] = 0.0f;
}

// ---------------- edge MLP (fused gather -> MLP -> scatter-add)
// 64 threads = 2 waves, 64 edges/block, each wave owns a 32-row M-tile (2 WMMA M-tiles)
// so every B fragment is reused twice.
__global__ __launch_bounds__(64) void edge_mlp_kernel(
    const float* __restrict__ h, const float* __restrict__ ea,
    const int* __restrict__ ei,
    const unsigned short* __restrict__ w1p, const unsigned short* __restrict__ w2p,
    const float* __restrict__ b1, const float* __restrict__ b2,
    float* __restrict__ agg) {
    __shared__ alignas(16) unsigned short Xs[64 * 160];   // 20 KB
    __shared__ alignas(16) unsigned short Ys[64 * 256];   // 32 KB
    const int tid = threadIdx.x;
    const int lane = tid & 31;
    const int wave = tid >> 5;            // 0..1
    const int e0 = blockIdx.x * 64;
    const int* srcI = ei;
    const int* dstI = ei + NE;

    // Gather [h[dst] | h[src] | edge_attr] -> bf16 tile (K = 160 = 5*32), float4 at a time
    for (int i = tid; i < 64 * 40; i += 64) {
        int el = i / 40, c4 = i - el * 40;
        int c = c4 * 4;
        int e = e0 + el;
        float4 v = make_float4(0.f, 0.f, 0.f, 0.f);
        if (e < NE) {
            if (c < 64)       v = *(const float4*)&h[(size_t)dstI[e] * DM + c];
            else if (c < 128) v = *(const float4*)&h[(size_t)srcI[e] * DM + (c - 64)];
            else              v = *(const float4*)&ea[(size_t)e * DE + (c - 128)];
        }
        us4 p = { f2bf(v.x), f2bf(v.y), f2bf(v.z), f2bf(v.w) };
        *(us4*)&Xs[el * 160 + c] = p;
    }
    __syncthreads();

    const int r0 = wave * 32;             // 32 rows per wave
    const int ncol = lane & 15;

    // GEMM1: [32,160] x [160,256] + b1, SiLU -> Ys (bf16)
    v16bf afr[2][5];
#pragma unroll
    for (int mt = 0; mt < 2; ++mt)
#pragma unroll
        for (int kt = 0; kt < 5; ++kt)
            afr[mt][kt] = load_a_frag(&Xs[(r0 + mt * 16) * 160 + kt * 32], 160, lane);

    for (int nt = 0; nt < 16; ++nt) {
        v8f acc0 = {}, acc1 = {};
#pragma unroll
        for (int kt = 0; kt < 5; ++kt) {
            v16bf b = load_b_frag(w1p, kt, nt, 16, lane);
            acc0 = wmma_bf16(afr[0][kt], b, acc0);
            acc1 = wmma_bf16(afr[1][kt], b, acc1);
        }
        int col = nt * 16 + ncol;
        float bias = b1[col];
#pragma unroll
        for (int i = 0; i < 8; ++i) {
            int rl = (lane < 16) ? i : (8 + i);
            Ys[(r0 + rl) * 256 + col]      = f2bf(silu(acc0[i] + bias));
            Ys[(r0 + 16 + rl) * 256 + col] = f2bf(silu(acc1[i] + bias));
        }
    }
    // each wave consumes only its own rows of Ys -> no barrier needed

    // GEMM2: [32,256] x [256,64] + b2, scatter-add into agg[dst]
#pragma unroll
    for (int nt = 0; nt < 4; ++nt) {
        v8f acc0 = {}, acc1 = {};
#pragma unroll
        for (int kt = 0; kt < 8; ++kt) {
            v16bf a0 = load_a_frag(&Ys[r0 * 256 + kt * 32], 256, lane);
            v16bf a1 = load_a_frag(&Ys[(r0 + 16) * 256 + kt * 32], 256, lane);
            v16bf b  = load_b_frag(w2p, kt, nt, 4, lane);
            acc0 = wmma_bf16(a0, b, acc0);
            acc1 = wmma_bf16(a1, b, acc1);
        }
        int col = nt * 16 + ncol;
        float bias = b2[col];
#pragma unroll
        for (int i = 0; i < 8; ++i) {
            int rl = (lane < 16) ? i : (8 + i);
            int ea0 = e0 + r0 + rl;
            int ea1 = ea0 + 16;
            if (ea0 < NE)
                atomicAdd(&agg[(size_t)dstI[ea0] * DM + col], acc0[i] + bias);
            if (ea1 < NE)
                atomicAdd(&agg[(size_t)dstI[ea1] * DM + col], acc1[i] + bias);
        }
    }
}

// ---------------- node update MLP: hp = h + MLP([h, agg])
__global__ __launch_bounds__(64) void node_mlp_kernel(
    const float* __restrict__ h, const float* __restrict__ agg,
    const unsigned short* __restrict__ u1p, const unsigned short* __restrict__ u2p,
    const float* __restrict__ b1, const float* __restrict__ b2,
    float* __restrict__ hp) {
    __shared__ alignas(16) unsigned short Xs[64 * 128];   // 16 KB
    __shared__ alignas(16) unsigned short Ys[64 * 256];   // 32 KB
    const int tid = threadIdx.x;
    const int lane = tid & 31;
    const int wave = tid >> 5;
    const int n0 = blockIdx.x * 64;

    for (int i = tid; i < 64 * 32; i += 64) {
        int nl = i >> 5, c4 = i & 31;
        int c = c4 * 4;
        int n = n0 + nl;
        float4 v = make_float4(0.f, 0.f, 0.f, 0.f);
        if (n < NN) {
            if (c < 64) v = *(const float4*)&h[(size_t)n * DM + c];
            else        v = *(const float4*)&agg[(size_t)n * DM + (c - 64)];
        }
        us4 p = { f2bf(v.x), f2bf(v.y), f2bf(v.z), f2bf(v.w) };
        *(us4*)&Xs[nl * 128 + c] = p;
    }
    __syncthreads();

    const int r0 = wave * 32;
    const int ncol = lane & 15;

    v16bf afr[2][4];
#pragma unroll
    for (int mt = 0; mt < 2; ++mt)
#pragma unroll
        for (int kt = 0; kt < 4; ++kt)
            afr[mt][kt] = load_a_frag(&Xs[(r0 + mt * 16) * 128 + kt * 32], 128, lane);

    for (int nt = 0; nt < 16; ++nt) {
        v8f acc0 = {}, acc1 = {};
#pragma unroll
        for (int kt = 0; kt < 4; ++kt) {
            v16bf b = load_b_frag(u1p, kt, nt, 16, lane);
            acc0 = wmma_bf16(afr[0][kt], b, acc0);
            acc1 = wmma_bf16(afr[1][kt], b, acc1);
        }
        int col = nt * 16 + ncol;
        float bias = b1[col];
#pragma unroll
        for (int i = 0; i < 8; ++i) {
            int rl = (lane < 16) ? i : (8 + i);
            Ys[(r0 + rl) * 256 + col]      = f2bf(silu(acc0[i] + bias));
            Ys[(r0 + 16 + rl) * 256 + col] = f2bf(silu(acc1[i] + bias));
        }
    }

#pragma unroll
    for (int nt = 0; nt < 4; ++nt) {
        v8f acc0 = {}, acc1 = {};
#pragma unroll
        for (int kt = 0; kt < 8; ++kt) {
            v16bf a0 = load_a_frag(&Ys[r0 * 256 + kt * 32], 256, lane);
            v16bf a1 = load_a_frag(&Ys[(r0 + 16) * 256 + kt * 32], 256, lane);
            v16bf b  = load_b_frag(u2p, kt, nt, 4, lane);
            acc0 = wmma_bf16(a0, b, acc0);
            acc1 = wmma_bf16(a1, b, acc1);
        }
        int col = nt * 16 + ncol;
        float bias = b2[col];
#pragma unroll
        for (int i = 0; i < 8; ++i) {
            int rl = (lane < 16) ? i : (8 + i);
            int na = n0 + r0 + rl;
            int nb = na + 16;
            if (na < NN)
                hp[(size_t)na * DM + col] = h[(size_t)na * DM + col] + acc0[i] + bias;
            if (nb < NN)
                hp[(size_t)nb * DM + col] = h[(size_t)nb * DM + col] + acc1[i] + bias;
        }
    }
}

// ---------------- GraphNorm: per-graph sums & counts
__global__ void gn_stats1(const float* __restrict__ hp, const int* __restrict__ batch,
                          float* __restrict__ sums, float* __restrict__ cnts) {
    int stride = gridDim.x * blockDim.x;
    for (int i = blockIdx.x * blockDim.x + threadIdx.x; i < NN * DM; i += stride) {
        int n = i >> 6, c = i & 63;
        int g = batch[n];
        atomicAdd(&sums[g * DM + c], hp[i]);
        if (c == 0) atomicAdd(&cnts[g], 1.0f);
    }
}

// ---------------- GraphNorm: center + variance accumulation
__global__ void gn_stats2(const float* __restrict__ hp, const int* __restrict__ batch,
                          const float* __restrict__ sums, const float* __restrict__ cnts,
                          const float* __restrict__ alpha,
                          float* __restrict__ hc, float* __restrict__ vars) {
    int stride = gridDim.x * blockDim.x;
    for (int i = blockIdx.x * blockDim.x + threadIdx.x; i < NN * DM; i += stride) {
        int n = i >> 6, c = i & 63;
        int g = batch[n];
        float rcnt = 1.0f / fmaxf(cnts[g], 1.0f);
        float mean = sums[g * DM + c] * rcnt;
        float v = hp[i] - alpha[c] * mean;
        hc[i] = v;
        atomicAdd(&vars[g * DM + c], v * v);
    }
}

// ---------------- GraphNorm: normalize
__global__ void gn_final(const float* __restrict__ hc, const int* __restrict__ batch,
                         const float* __restrict__ vars, const float* __restrict__ cnts,
                         const float* __restrict__ w, const float* __restrict__ b,
                         float* __restrict__ out) {
    int stride = gridDim.x * blockDim.x;
    for (int i = blockIdx.x * blockDim.x + threadIdx.x; i < NN * DM; i += stride) {
        int n = i >> 6, c = i & 63;
        int g = batch[n];
        float rcnt = 1.0f / fmaxf(cnts[g], 1.0f);
        float var = vars[g * DM + c] * rcnt;
        out[i] = w[c] * hc[i] * rsqrtf(var + EPSV) + b[c];
    }
}

extern "C" void kernel_launch(void* const* d_in, const int* in_sizes, int n_in,
                              void* d_out, int out_size, void* d_ws, size_t ws_size,
                              hipStream_t stream) {
    (void)in_sizes; (void)n_in; (void)out_size; (void)ws_size;
    const float* h    = (const float*)d_in[0];
    const float* ea   = (const float*)d_in[1];
    const int*   ei   = (const int*)d_in[2];
    const int*   bat  = (const int*)d_in[3];
    const float* mw1  = (const float*)d_in[4];
    const float* mb1  = (const float*)d_in[5];
    const float* mw2  = (const float*)d_in[6];
    const float* mb2  = (const float*)d_in[7];
    const float* uw1  = (const float*)d_in[8];
    const float* ub1  = (const float*)d_in[9];
    const float* uw2  = (const float*)d_in[10];
    const float* ub2  = (const float*)d_in[11];
    const float* gw   = (const float*)d_in[12];
    const float* gb   = (const float*)d_in[13];
    const float* ga   = (const float*)d_in[14];
    float* out = (float*)d_out;

    char* ws = (char*)d_ws;
    size_t off = 0;
    auto take = [&](size_t bytes) -> void* {
        void* p = ws + off;
        off += (bytes + 255) & ~(size_t)255;
        return p;
    };
    unsigned short* w1p = (unsigned short*)take((size_t)5 * 16 * 32 * 16 * 2);
    unsigned short* w2p = (unsigned short*)take((size_t)8 * 4 * 32 * 16 * 2);
    unsigned short* u1p = (unsigned short*)take((size_t)4 * 16 * 32 * 16 * 2);
    unsigned short* u2p = (unsigned short*)take((size_t)8 * 4 * 32 * 16 * 2);
    float* agg  = (float*)take((size_t)NN * DM * 4);
    float* hp   = (float*)take((size_t)NN * DM * 4);
    float* hc   = (float*)take((size_t)NN * DM * 4);
    float* sums = (float*)take((size_t)NG * DM * 4);
    float* vars = (float*)take((size_t)NG * DM * 4);
    float* cnts = (float*)take((size_t)NG * 4);

    zero_ws<<<4096, 256, 0, stream>>>(agg, sums, vars, cnts);
    pack_weights<<<10, 256, 0, stream>>>(mw1, 160, 256, w1p);
    pack_weights<<<4,  256, 0, stream>>>(mw2, 256, 64,  w2p);
    pack_weights<<<8,  256, 0, stream>>>(uw1, 128, 256, u1p);
    pack_weights<<<4,  256, 0, stream>>>(uw2, 256, 64,  u2p);

    edge_mlp_kernel<<<(NE + 63) / 64, 64, 0, stream>>>(h, ea, ei, w1p, w2p, mb1, mb2, agg);
    node_mlp_kernel<<<(NN + 63) / 64, 64, 0, stream>>>(h, agg, u1p, u2p, ub1, ub2, hp);

    gn_stats1<<<4096, 256, 0, stream>>>(hp, bat, sums, cnts);
    gn_stats2<<<4096, 256, 0, stream>>>(hp, bat, sums, cnts, ga, hc, vars);
    gn_final<<<4096, 256, 0, stream>>>(hc, bat, vars, cnts, gw, gb, out);
}
